// MultiheadAttention_45896020525620
// MI455X (gfx1250) — matrix operand order repaired and checked by reading, not compiled, not run on hardware
//
#include <hip/hip_runtime.h>
#include <math.h>
#include <stdint.h>

// ---------------------------------------------------------------------------
// CDNA5 (gfx1250) fused prefix-tuned multihead attention, bf16 WMMA pipeline.
//   1) qkv  = query @ Win^T + bin          (GEMM 16384x3072x1024, bf16 out)
//   2) attn = softmax(qk^T)*v + w_b*softmax(q pk^T)*pv   (WMMA flash-attn)
//   3) out  = attn @ Wout^T + bout         (GEMM 16384x1024x1024, f32 out)
// Round 3: software-pipelined staging. Global fetches (regs or async-to-LDS)
// for tile t+1 are issued before the compute of tile t, so LOADcnt/ASYNCcnt
// drain under the WMMA block instead of stalling the wave.
// ---------------------------------------------------------------------------

typedef __attribute__((ext_vector_type(16))) __bf16 v16bf;
typedef __attribute__((ext_vector_type(8)))  float  v8f;

struct __align__(16) bfoct { __bf16 e[8]; };
union Frag16 { v16bf v; bfoct o[2]; __bf16 e[16]; };

static __device__ __forceinline__ v8f wmma_bf16(v16bf a, v16bf b, v8f c) {
  return __builtin_amdgcn_wmma_f32_16x16x32_bf16(false, a, false, b, (short)0, c,
                                                 false, false);
}

// ---- 16-lane butterfly reductions (VALU v_permlane16_b32) -----------------
#if __has_builtin(__builtin_amdgcn_permlane16)
static __device__ __forceinline__ float pl16(float x, unsigned s0, unsigned s1) {
  return __int_as_float(__builtin_amdgcn_permlane16(
      __float_as_int(x), __float_as_int(x), s0, s1, false, false));
}
static __device__ __forceinline__ float rmax16(float x) {
  x = fmaxf(x, pl16(x, 0x67452301u, 0xEFCDAB89u));  // xor 1
  x = fmaxf(x, pl16(x, 0x54761032u, 0xDCFE98BAu));  // xor 2
  x = fmaxf(x, pl16(x, 0x32107654u, 0xBA98FEDCu));  // xor 4
  x = fmaxf(x, pl16(x, 0xFEDCBA98u, 0x76543210u));  // xor 8
  return x;
}
static __device__ __forceinline__ float rsum16(float x) {
  x += pl16(x, 0x67452301u, 0xEFCDAB89u);
  x += pl16(x, 0x54761032u, 0xDCFE98BAu);
  x += pl16(x, 0x32107654u, 0xBA98FEDCu);
  x += pl16(x, 0xFEDCBA98u, 0x76543210u);
  return x;
}
#else
static __device__ __forceinline__ float rmax16(float x) {
  for (int off = 1; off < 16; off <<= 1) x = fmaxf(x, __shfl_xor(x, off, 32));
  return x;
}
static __device__ __forceinline__ float rsum16(float x) {
  for (int off = 1; off < 16; off <<= 1) x += __shfl_xor(x, off, 32);
  return x;
}
#endif

// ---- staging helpers ------------------------------------------------------
static __device__ __forceinline__ bfoct cvt8(float4 a, float4 b) {
  bfoct o;
  o.e[0] = (__bf16)a.x; o.e[1] = (__bf16)a.y;
  o.e[2] = (__bf16)a.z; o.e[3] = (__bf16)a.w;
  o.e[4] = (__bf16)b.x; o.e[5] = (__bf16)b.y;
  o.e[6] = (__bf16)b.z; o.e[7] = (__bf16)b.w;
  return o;
}
static __device__ __forceinline__ bfoct cvt_oct(const float* p) {
  return cvt8(*(const float4*)p, *(const float4*)(p + 4));
}
// bf16 source: async DMA global -> LDS, tracked by ASYNCcnt
static __device__ __forceinline__ void stage_async(bfoct* dst, const __bf16* src) {
  unsigned lds = (unsigned)(uintptr_t)dst;            // low 32b = LDS offset
  unsigned long long ga = (unsigned long long)(uintptr_t)src;
  asm volatile("global_load_async_to_lds_b128 %0, %1, off"
               :: "v"(lds), "v"(ga) : "memory");
}
static __device__ __forceinline__ void wait_async() {
  asm volatile("s_wait_asynccnt 0x0" ::: "memory");
}

// ---------------------------------------------------------------------------
// C[M,N] = A[M,K] @ W[N,K]^T + bias[N].  BM=128 BN=128 BK=32, 8 waves/WG,
// each wave computes a 32x64 tile as 2x4 WMMA fragments (8 WMMA / k-step).
// W (f32) is reg-pipelined one k-step ahead; A is reg-pipelined (f32) or
// double-buffered async-to-LDS (bf16).
// ---------------------------------------------------------------------------
template <typename AT, typename OT>
__global__ __launch_bounds__(256) void gemm_bt(const AT* __restrict__ A,
                                               const float* __restrict__ W,
                                               const float* __restrict__ bias,
                                               OT* __restrict__ C,
                                               int M, int N, int K) {
  constexpr bool ABF = (sizeof(AT) == 2);
  __shared__ bfoct As[ABF ? 2 : 1][128][5];  // [m][k-oct], 4 used + 1 pad
  __shared__ bfoct Bs[128][5];               // [n][k-oct]
  const int tid  = threadIdx.x;
  const int lane = tid & 31, w = tid >> 5;
  const int ha = lane >> 4, l16 = lane & 15;
  const int m0 = blockIdx.y * 128, n0 = blockIdx.x * 128;
  const int mw = (w >> 1) * 32, nw = (w & 1) * 64;
  const int r0 = tid >> 2, c0 = tid & 3;  // this thread's oct slots: r0, r0+64
  v8f acc[2][4] = {};

  float4 wreg[4], areg[4];
  auto loadW = [&](int kb) {
    const float* p0 = W + (size_t)(n0 + r0) * K + kb + c0 * 8;
    const float* p1 = W + (size_t)(n0 + r0 + 64) * K + kb + c0 * 8;
    wreg[0] = *(const float4*)p0; wreg[1] = *(const float4*)(p0 + 4);
    wreg[2] = *(const float4*)p1; wreg[3] = *(const float4*)(p1 + 4);
  };
  auto stageA = [&](int buf, int kb) {
    if constexpr (ABF) {
      stage_async(&As[buf][r0][c0],
                  A + (size_t)(m0 + r0) * K + kb + c0 * 8);
      stage_async(&As[buf][r0 + 64][c0],
                  A + (size_t)(m0 + r0 + 64) * K + kb + c0 * 8);
    } else {
      (void)buf;
      const float* p0 = A + (size_t)(m0 + r0) * K + kb + c0 * 8;
      const float* p1 = A + (size_t)(m0 + r0 + 64) * K + kb + c0 * 8;
      areg[0] = *(const float4*)p0; areg[1] = *(const float4*)(p0 + 4);
      areg[2] = *(const float4*)p1; areg[3] = *(const float4*)(p1 + 4);
    }
  };

  // prologue: fetch tile 0
  loadW(0);
  stageA(0, 0);

  const int nst = K / 32;
  for (int s = 0; s < nst; ++s) {
    // store-phase: regs (tile s) -> LDS
    Bs[r0][c0]      = cvt8(wreg[0], wreg[1]);
    Bs[r0 + 64][c0] = cvt8(wreg[2], wreg[3]);
    if constexpr (!ABF) {
      As[0][r0][c0]      = cvt8(areg[0], areg[1]);
      As[0][r0 + 64][c0] = cvt8(areg[2], areg[3]);
    }
    wait_async();
    __syncthreads();
    // issue fetches for tile s+1 (drain during compute below)
    if (s + 1 < nst) {
      loadW((s + 1) * 32);
      stageA(ABF ? ((s + 1) & 1) : 0, (s + 1) * 32);
    }
    // compute tile s
    const bfoct(*Ac)[5] = As[ABF ? (s & 1) : 0];
    Frag16 a[2], bfr[4];
#pragma unroll
    for (int mt = 0; mt < 2; ++mt) {
      int r = mw + mt * 16 + l16;
      a[mt].o[0] = Ac[r][ha];          // K = ha*8 + 0..7
      a[mt].o[1] = Ac[r][2 + ha];      // K = 16 + ha*8 + 0..7
    }
#pragma unroll
    for (int nt = 0; nt < 4; ++nt) {
      int r = nw + nt * 16 + l16;
      bfr[nt].o[0] = Bs[r][ha];          // K = ha*8 + 0..7
      bfr[nt].o[1] = Bs[r][2 + ha];      // K = 16 + ha*8 + 0..7
    }
#pragma unroll
    for (int mt = 0; mt < 2; ++mt)
#pragma unroll
      for (int nt = 0; nt < 4; ++nt)
        acc[mt][nt] = wmma_bf16(a[mt].v, bfr[nt].v, acc[mt][nt]);
    __syncthreads();
  }

#pragma unroll
  for (int mt = 0; mt < 2; ++mt)
#pragma unroll
    for (int nt = 0; nt < 4; ++nt) {
      int col = n0 + nw + nt * 16 + l16;
      float bv = bias[col];
#pragma unroll
      for (int i = 0; i < 8; ++i) {
        int row = m0 + mw + mt * 16 + ha * 8 + i;
        C[(size_t)row * N + col] = (OT)(acc[mt][nt][i] + bv);
      }
    }
}

// ---------------------------------------------------------------------------
// Attention: grid (S/128, H, B); 8 waves x 16 query rows. Online softmax over
// 16 key tiles of 32 (K async double-buffered, V reg-pipelined), then a
// separate 16-key prefix softmax weighted by batch_weight.
// ---------------------------------------------------------------------------
#define S_  512
#define B_  32
#define E_  1024
#define H_  16
#define D_  64
#define L_  16
#define N3_ 3072

__global__ __launch_bounds__(256) void attn_kernel(
    const __bf16* __restrict__ qkv, const float* __restrict__ prefix_pool,
    const int* __restrict__ prompt_ids, const float* __restrict__ batch_weight,
    __bf16* __restrict__ attn_out) {
  __shared__ bfoct Ks[2][32][9];  // [key][d-oct], 8 used + 1 pad
  __shared__ bfoct Vt[64][5];     // [d][key-oct], 4 used + 1 pad
  __shared__ bfoct Pb[8][16][5];  // per-wave probs [row][key-oct]
  const int tid = threadIdx.x, lane = tid & 31, w = tid >> 5;
  const int ha = lane >> 4, l16 = lane & 15;
  const int qb = blockIdx.x * 128 + w * 16;
  const int h = blockIdx.y, b = blockIdx.z;
  const float scale = 0.125f;  // 1/sqrt(64), applied to scores post-WMMA

  // Q fragments (A layout), rows qb..qb+15, K-chunks d=[0,32) and [32,64)
  Frag16 qf[2];
  {
    const __bf16* qrow = qkv + ((size_t)(qb + l16) * B_ + b) * N3_ + h * D_;
#pragma unroll
    for (int kk = 0; kk < 2; ++kk) {
      qf[kk].o[0] = *(const bfoct*)(qrow + kk * 32 + ha * 8);
      qf[kk].o[1] = *(const bfoct*)(qrow + kk * 32 + 16 + ha * 8);
    }
  }

  float rm[8], rl[8];
  v8f o[4] = {};
#pragma unroll
  for (int i = 0; i < 8; ++i) { rm[i] = -1e30f; rl[i] = 0.0f; }
  __bf16* pbw = (__bf16*)&Pb[w][0][0];  // row stride 40 bf16
  __bf16* vte = (__bf16*)&Vt[0][0];     // row stride 40 bf16

  const int key = tid >> 3, oc = tid & 7;
  auto kvBase = [&](int it) {
    return qkv + ((size_t)(it * 32 + key) * B_ + b) * N3_ + h * D_ + oc * 8;
  };

  // prologue: async K tile 0, V tile 0 into regs
  stage_async(&Ks[0][key][oc], kvBase(0) + E_);
  bfoct vo = *(const bfoct*)(kvBase(0) + 2 * E_);

  for (int it = 0; it < S_ / 32; ++it) {
    // store-phase: transpose V tile (regs from previous fetch) into Vt
#pragma unroll
    for (int q = 0; q < 8; ++q) vte[(oc * 8 + q) * 40 + key] = vo.e[q];
    wait_async();
    __syncthreads();
    // issue fetches for tile it+1 (drain during compute below)
    if (it + 1 < S_ / 32) {
      stage_async(&Ks[(it + 1) & 1][key][oc], kvBase(it + 1) + E_);
      vo = *(const bfoct*)(kvBase(it + 1) + 2 * E_);
    }

    const bfoct(*Kc)[9] = Ks[it & 1];
    v8f sc0 = {}, sc1 = {};  // scores: 16 queries x keys [0,16) and [16,32)
#pragma unroll
    for (int kk = 0; kk < 2; ++kk) {
      Frag16 k0, k1;
      k0.o[0] = Kc[l16][kk * 4 + ha];
      k0.o[1] = Kc[l16][kk * 4 + 2 + ha];
      k1.o[0] = Kc[16 + l16][kk * 4 + ha];
      k1.o[1] = Kc[16 + l16][kk * 4 + 2 + ha];
      sc0 = wmma_bf16(qf[kk].v, k0.v, sc0);
      sc1 = wmma_bf16(qf[kk].v, k1.v, sc1);
    }

    // online softmax (row = ha*8+i spread over the 16 lanes of each half)
#pragma unroll
    for (int i = 0; i < 8; ++i) {
      float s0 = sc0[i] * scale, s1 = sc1[i] * scale;
      float mx = rmax16(fmaxf(s0, s1));
      float nm = fmaxf(rm[i], mx);
      float f = __expf(rm[i] - nm);
      rm[i] = nm;
      float p0 = __expf(s0 - nm), p1 = __expf(s1 - nm);
      pbw[(ha * 8 + i) * 40 + l16] = (__bf16)p0;
      pbw[(ha * 8 + i) * 40 + 16 + l16] = (__bf16)p1;
      rl[i] = rl[i] * f + rsum16(p0 + p1);
#pragma unroll
      for (int j = 0; j < 4; ++j) o[j][i] *= f;
    }

    // PV: probs(16x32) @ v(32x64)
    Frag16 pa;
    pa.o[0] = Pb[w][l16][ha];
    pa.o[1] = Pb[w][l16][2 + ha];
#pragma unroll
    for (int j = 0; j < 4; ++j) {
      Frag16 vf;
      vf.o[0] = Vt[j * 16 + l16][ha];
      vf.o[1] = Vt[j * 16 + l16][2 + ha];
      o[j] = wmma_bf16(pa.v, vf.v, o[j]);
    }
    __syncthreads();
  }

  // ---- prefix attention: independent softmax over L=16 keys ----
  const int pid = prompt_ids[b];
  const float wgt = batch_weight[b];
  {
    const float* pkbase = prefix_pool + (size_t)pid * 2 * L_ * E_ + h * D_;
    const float* pvbase = pkbase + (size_t)L_ * E_;
    if (tid < 128) {  // pk -> Ks[0] rows 0..15 (f32 -> bf16 convert)
      int l = tid >> 3, pc = tid & 7;
      Ks[0][l][pc] = cvt_oct(pkbase + (size_t)l * E_ + pc * 8);
    }
    {  // pv -> Vt transposed; keys 16..31 zeroed
      bfoct po_;
      if (key < L_) {
        po_ = cvt_oct(pvbase + (size_t)key * E_ + oc * 8);
      } else {
#pragma unroll
        for (int q = 0; q < 8; ++q) po_.e[q] = (__bf16)0.0f;
      }
#pragma unroll
      for (int q = 0; q < 8; ++q) vte[(oc * 8 + q) * 40 + key] = po_.e[q];
    }
  }
  __syncthreads();

  v8f sp = {};
#pragma unroll
  for (int kk = 0; kk < 2; ++kk) {
    Frag16 kf;
    kf.o[0] = Ks[0][l16][kk * 4 + ha];
    kf.o[1] = Ks[0][l16][kk * 4 + 2 + ha];
    sp = wmma_bf16(qf[kk].v, kf.v, sp);
  }
  float pl[8];
#pragma unroll
  for (int i = 0; i < 8; ++i) {
    float s = sp[i] * scale;
    float mx = rmax16(s);
    float p = __expf(s - mx);
    pbw[(ha * 8 + i) * 40 + l16] = (__bf16)p;
    pbw[(ha * 8 + i) * 40 + 16 + l16] = (__bf16)0.0f;
    pl[i] = rsum16(p);
  }
  v8f po[4] = {};
  {
    Frag16 pa;
    pa.o[0] = Pb[w][l16][ha];
    pa.o[1] = Pb[w][l16][2 + ha];
#pragma unroll
    for (int j = 0; j < 4; ++j) {
      Frag16 vf;
      vf.o[0] = Vt[j * 16 + l16][ha];
      vf.o[1] = Vt[j * 16 + l16][2 + ha];
      po[j] = wmma_bf16(pa.v, vf.v, po[j]);
    }
  }

  // combine and store: row s = qb + ha*8 + i, col = h*64 + j*16 + l16
#pragma unroll
  for (int i = 0; i < 8; ++i) {
    float invl = 1.0f / rl[i];
    float invp = wgt / pl[i];
    int s = qb + ha * 8 + i;
    size_t base = ((size_t)s * B_ + b) * E_ + h * D_ + l16;
#pragma unroll
    for (int j = 0; j < 4; ++j)
      attn_out[base + j * 16] = (__bf16)(o[j][i] * invl + po[j][i] * invp);
  }
}

// ---------------------------------------------------------------------------
extern "C" void kernel_launch(void* const* d_in, const int* in_sizes, int n_in,
                              void* d_out, int out_size, void* d_ws,
                              size_t ws_size, hipStream_t stream) {
  (void)in_sizes; (void)n_in; (void)out_size; (void)ws_size;
  const float* query         = (const float*)d_in[0];
  const int*   prompt_ids    = (const int*)d_in[1];
  const float* batch_weight  = (const float*)d_in[2];
  const float* in_proj_w     = (const float*)d_in[3];
  const float* in_proj_b     = (const float*)d_in[4];
  const float* out_proj_w    = (const float*)d_in[5];
  const float* out_proj_b    = (const float*)d_in[6];
  const float* prefix_pool   = (const float*)d_in[7];

  const int M = S_ * B_;  // 16384 rows
  __bf16* qkv     = (__bf16*)d_ws;                     // [M][3072] bf16 (96MB)
  __bf16* attnbuf = qkv + (size_t)M * N3_;             // [M][1024] bf16 (32MB)
  float*  out     = (float*)d_out;

  gemm_bt<float, __bf16><<<dim3(N3_ / 128, M / 128), 256, 0, stream>>>(
      query, in_proj_w, in_proj_b, qkv, M, N3_, E_);
  attn_kernel<<<dim3(S_ / 128, H_, B_), 256, 0, stream>>>(
      qkv, prefix_pool, prompt_ids, batch_weight, attnbuf);
  gemm_bt<__bf16, float><<<dim3(E_ / 128, M / 128), 256, 0, stream>>>(
      attnbuf, out_proj_w, out_proj_b, out, M, E_, E_);
}
